// SigmoidMoERouter_63015760167161
// MI455X (gfx1250) — matrix operand order repaired and checked by reading, not compiled, not run on hardware
//
#include <hip/hip_runtime.h>
#include <hip/hip_bf16.h>

// Problem constants (match reference)
#define D_DIM   1024
#define T_TOK   2048
#define NEXP    8
#define EDIM    2048
#define HALF_H  1024          // process H in two phases to halve LDS
#define TILE_M  32            // tokens per expert tile (2 x WMMA M)
#define XS      1032          // padded LDS row stride (bf16 elems): 2064B -> conflict-free
#define HS      1032

typedef __attribute__((ext_vector_type(16))) __bf16 v16bf;
typedef __attribute__((ext_vector_type(8)))  float  v8f;

// fast sigmoid via hardware v_rcp_f32 (plenty accurate for bf16 storage)
__device__ __forceinline__ float fast_sigmoid(float v) {
  return __builtin_amdgcn_rcpf(1.f + __expf(-v));
}

// ---- fragment helpers -------------------------------------------------------

// A-fragment (16x32 bf16) from LDS, contiguous K order (matches B):
// lanes 0-15 hold K = 0..15 (kq=0); lanes 16-31 hold K = 16..31 (kq=8).
__device__ __forceinline__ v16bf lds_frag_a(const __bf16* rowp, int k0, int kq) {
  union { uint4 q[2]; v16bf v; } u;
  u.q[0] = *(const uint4*)(rowp + k0 + 2 * kq);
  u.q[1] = *(const uint4*)(rowp + k0 + 2 * kq + 8);
  return u.v;
}

// B-fragment from pre-converted bf16 weights: 16 contiguous bf16 = 2 x b128.
__device__ __forceinline__ v16bf load_frag_b16(const __bf16* p) {
  union { uint4 q[2]; v16bf v; } u;
  u.q[0] = ((const uint4*)p)[0];
  u.q[1] = ((const uint4*)p)[1];
  return u.v;
}

// B-fragment from f32 weights with on-the-fly conversion (fallback path).
__device__ __forceinline__ v16bf load_frag_bf32(const float* p) {
  const float4* p4 = (const float4*)p;
  float4 f0 = p4[0], f1 = p4[1], f2 = p4[2], f3 = p4[3];
  union { __bf16 b[16]; v16bf v; } u;
  u.b[0]  = (__bf16)f0.x; u.b[1]  = (__bf16)f0.y; u.b[2]  = (__bf16)f0.z; u.b[3]  = (__bf16)f0.w;
  u.b[4]  = (__bf16)f1.x; u.b[5]  = (__bf16)f1.y; u.b[6]  = (__bf16)f1.z; u.b[7]  = (__bf16)f1.w;
  u.b[8]  = (__bf16)f2.x; u.b[9]  = (__bf16)f2.y; u.b[10] = (__bf16)f2.z; u.b[11] = (__bf16)f2.w;
  u.b[12] = (__bf16)f3.x; u.b[13] = (__bf16)f3.y; u.b[14] = (__bf16)f3.z; u.b[15] = (__bf16)f3.w;
  return u.v;
}

#define WMMA_BF16(a, b, c) \
  __builtin_amdgcn_wmma_f32_16x16x32_bf16(false, (a), false, (b), (short)0, (c), false, false)

// ---- kernel 0: zero output + per-expert counters ----------------------------
__global__ __launch_bounds__(256) void zero_kernel(float4* __restrict__ out4, int n4,
                                                   int* __restrict__ counts) {
  int i = blockIdx.x * blockDim.x + threadIdx.x;
  if (i < n4) out4[i] = make_float4(0.f, 0.f, 0.f, 0.f);
  if (i < NEXP) counts[i] = 0;
}

// ---- kernel 0b: f32 -> bf16 weight conversion -------------------------------
__global__ __launch_bounds__(256) void convert_bf16_kernel(const float4* __restrict__ in,
                                                           uint2* __restrict__ out, int n4) {
  int i = blockIdx.x * blockDim.x + threadIdx.x;
  if (i < n4) {
    float4 v = in[i];
    union { __bf16 b[4]; uint2 u; } pk;
    pk.b[0] = (__bf16)v.x; pk.b[1] = (__bf16)v.y;
    pk.b[2] = (__bf16)v.z; pk.b[3] = (__bf16)v.w;
    out[i] = pk.u;
  }
}

// ---- kernel 1: gating (one wave per token) ----------------------------------
__global__ __launch_bounds__(256) void gate_kernel(const float* __restrict__ x,
                                                   const float* __restrict__ gw,
                                                   const float* __restrict__ bias,
                                                   float* __restrict__ comb,
                                                   int* __restrict__ counts,
                                                   int* __restrict__ idx) {
  const int lane = threadIdx.x & 31;
  const int wv   = threadIdx.x >> 5;
  const int t    = blockIdx.x * 8 + wv;

  float acc[NEXP];
#pragma unroll
  for (int e = 0; e < NEXP; ++e) acc[e] = 0.f;

  const float* xr = x + (size_t)t * D_DIM;
  for (int d = lane; d < D_DIM; d += 32) {
    float xv = xr[d];
#pragma unroll
    for (int e = 0; e < NEXP; ++e) acc[e] += xv * gw[e * D_DIM + d];
  }
#pragma unroll
  for (int e = 0; e < NEXP; ++e) {
    for (int off = 16; off > 0; off >>= 1)
      acc[e] += __shfl_xor(acc[e], off, 32);
  }

  if (lane == 0) {
    float s[NEXP];
#pragma unroll
    for (int e = 0; e < NEXP; ++e)
      s[e] = fast_sigmoid(acc[e] + bias[e]);
    int i1 = 0; float b1 = s[0];
#pragma unroll
    for (int e = 1; e < NEXP; ++e) if (s[e] > b1) { b1 = s[e]; i1 = e; }
    int i2 = (i1 == 0) ? 1 : 0; float b2 = s[i2];
#pragma unroll
    for (int e = 0; e < NEXP; ++e)
      if (e != i1 && s[e] > b2) { b2 = s[e]; i2 = e; }
    float denom = b1 + b2 + 1e-6f;
    float c1 = b1 / denom, c2 = b2 / denom;   // cold path: keep exact division
#pragma unroll
    for (int e = 0; e < NEXP; ++e) comb[t * NEXP + e] = 0.f;
    comb[t * NEXP + i1] = c1;
    comb[t * NEXP + i2] = c2;
    int p1 = atomicAdd(&counts[i1], 1); idx[i1 * T_TOK + p1] = t;
    int p2 = atomicAdd(&counts[i2], 1); idx[i2 * T_TOK + p2] = t;
  }
}

// ---- kernel 2: fused expert FFN (gather -> GEMM1 -> SiLU -> GEMM2 -> scatter)
// M = 32 tokens per block: each B fragment feeds two WMMAs (rows 0-15, 16-31).
template <bool PRECONV>
__global__ __launch_bounds__(256) void moe_expert_kernel(const float* __restrict__ x,
                                                         const __bf16* __restrict__ w1b,
                                                         const __bf16* __restrict__ w2b,
                                                         const float* __restrict__ w1f,
                                                         const float* __restrict__ w2f,
                                                         const float* __restrict__ comb,
                                                         const int* __restrict__ counts,
                                                         const int* __restrict__ idx,
                                                         float* __restrict__ out) {
  __shared__ __align__(16) __bf16 xs[TILE_M * XS];   // 66 KB: bf16 x tile
  __shared__ __align__(16) __bf16 hs[TILE_M * HS];   // 66 KB: bf16 silu(h) half
  __shared__ int   tokLds[TILE_M];
  __shared__ float scaleLds[TILE_M];

  const int e    = blockIdx.y;
  const int tile = blockIdx.x;
  const int cnt  = counts[e];
  const int row0 = tile * TILE_M;
  if (row0 >= cnt) return;
  const int rows = min(TILE_M, cnt - row0);

  const int tid  = threadIdx.x;
  const int lane = tid & 31;
  const int wv   = tid >> 5;
  const int nL   = lane & 15;   // fragment column (B/C) or row (A)
  const int hgrp = lane >> 4;   // lane half-group

  if (tid < TILE_M) {
    int r = tid;
    int tok = (r < rows) ? idx[e * T_TOK + row0 + r] : 0;
    tokLds[r]   = tok;
    scaleLds[r] = (r < rows) ? comb[tok * NEXP + e] : 0.f;
  }
  __syncthreads();

  // gather x rows -> bf16 LDS (zero-pad invalid rows so silu(0)=0 kills them)
  for (int i = tid; i < TILE_M * (D_DIM / 4); i += 256) {
    int r  = i >> 8;           // / (D_DIM/4)
    int c4 = i & 255;
    float4 v = make_float4(0.f, 0.f, 0.f, 0.f);
    if (r < rows) v = ((const float4*)(x + (size_t)tokLds[r] * D_DIM))[c4];
    union { __bf16 b[4]; uint2 u; } pk;
    pk.b[0] = (__bf16)v.x; pk.b[1] = (__bf16)v.y;
    pk.b[2] = (__bf16)v.z; pk.b[3] = (__bf16)v.w;
    *(uint2*)&xs[r * XS + c4 * 4] = pk.u;
  }
  __syncthreads();

  v8f acc2[8][2];               // persistent GEMM2 accumulators: 8 d-tiles x 2 m-groups
#pragma unroll
  for (int i = 0; i < 8; ++i) { acc2[i][0] = {}; acc2[i][1] = {}; }

  const __bf16* xrow0 = &xs[nL * XS];
  const __bf16* xrow1 = &xs[(nL + 16) * XS];
  const __bf16* hrow0 = &hs[nL * HS];
  const __bf16* hrow1 = &hs[(nL + 16) * HS];

  for (int ph = 0; ph < 2; ++ph) {
    const int hbase = ph * HALF_H;

    // ---- GEMM1: h[:, local 0..1023] = x(32x1024) * w1[e, hbase+j, :]^T, + SiLU
#pragma unroll 1
    for (int jt = 0; jt < 8; ++jt) {
      const int j0 = wv * 128 + jt * 16;      // local j of this 16-wide tile
      v8f acc0 = {}, acc1 = {};
      const size_t wro = ((size_t)e * EDIM + hbase + j0 + nL) * D_DIM + hgrp * 16;
#pragma unroll 2
      for (int kk = 0; kk < D_DIM; kk += 32) {
        v16bf b = PRECONV ? load_frag_b16(w1b + wro + kk)
                          : load_frag_bf32(w1f + wro + kk);
        v16bf a0 = lds_frag_a(xrow0, kk, hgrp * 8);
        v16bf a1 = lds_frag_a(xrow1, kk, hgrp * 8);
        acc0 = WMMA_BF16(a0, b, acc0);
        acc1 = WMMA_BF16(a1, b, acc1);
      }
#pragma unroll
      for (int v = 0; v < 8; ++v) {           // C layout: m = v + 8*hgrp, n = nL
        int m = v + hgrp * 8;
        float h0 = acc0[v], h1 = acc1[v];
        hs[m * HS + j0 + nL]        = (__bf16)(h0 * fast_sigmoid(h0));  // silu
        hs[(m + 16) * HS + j0 + nL] = (__bf16)(h1 * fast_sigmoid(h1));
      }
    }
    __syncthreads();

    // ---- GEMM2 (accumulate across phases): out += h(32x1024) * w2[e, d, hbase+k]^T
#pragma unroll
    for (int dt = 0; dt < 8; ++dt) {
      const int d0 = wv * 128 + dt * 16;
      v8f acc0 = acc2[dt][0], acc1 = acc2[dt][1];
      const size_t wro = ((size_t)e * D_DIM + d0 + nL) * EDIM + hbase + hgrp * 16;
#pragma unroll 2
      for (int kk = 0; kk < HALF_H; kk += 32) {
        v16bf b = PRECONV ? load_frag_b16(w2b + wro + kk)
                          : load_frag_bf32(w2f + wro + kk);
        v16bf a0 = lds_frag_a(hrow0, kk, hgrp * 8);
        v16bf a1 = lds_frag_a(hrow1, kk, hgrp * 8);
        acc0 = WMMA_BF16(a0, b, acc0);
        acc1 = WMMA_BF16(a1, b, acc1);
      }
      acc2[dt][0] = acc0; acc2[dt][1] = acc1;
    }
    __syncthreads();
  }

  // ---- scale by gate weight and scatter (exactly 2 experts/token -> atomicAdd
  //      is commutative => deterministic)
#pragma unroll
  for (int dt = 0; dt < 8; ++dt) {
#pragma unroll
    for (int v = 0; v < 8; ++v) {
      int m = v + hgrp * 8;
      int col = wv * 128 + dt * 16 + nL;
      if (m < rows)
        atomicAdd(&out[(size_t)tokLds[m] * D_DIM + col], acc2[dt][0][v] * scaleLds[m]);
      if (m + 16 < rows)
        atomicAdd(&out[(size_t)tokLds[m + 16] * D_DIM + col], acc2[dt][1][v] * scaleLds[m + 16]);
    }
  }
}

// ---- host entry -------------------------------------------------------------
extern "C" void kernel_launch(void* const* d_in, const int* in_sizes, int n_in,
                              void* d_out, int out_size, void* d_ws, size_t ws_size,
                              hipStream_t stream) {
  const float* x    = (const float*)d_in[0];   // [1,2048,1024]
  const float* gw   = (const float*)d_in[1];   // [8,1024]
  const float* bias = (const float*)d_in[2];   // [8]
  const float* w1   = (const float*)d_in[3];   // [8,2048,1024]
  const float* w2   = (const float*)d_in[4];   // [8,1024,2048]
  float* out        = (float*)d_out;           // [1,2048,1024]

  // workspace layout: comb[T*E] f32 | counts[64] i32 | idx[E*T] i32 | bf16 weights
  float* comb  = (float*)d_ws;
  int* counts  = (int*)((char*)d_ws + (size_t)T_TOK * NEXP * sizeof(float));
  int* idxList = counts + 64;
  const size_t wElems = (size_t)NEXP * EDIM * D_DIM;            // per weight matrix
  const size_t wOff   = (size_t)T_TOK * NEXP * 4 + 256 + (size_t)NEXP * T_TOK * 4;
  __bf16* w1b = (__bf16*)((char*)d_ws + wOff);
  __bf16* w2b = w1b + wElems;
  const bool preconv = ws_size >= wOff + 2 * wElems * sizeof(__bf16);

  const int n4 = T_TOK * D_DIM / 4;
  zero_kernel<<<(n4 + 255) / 256, 256, 0, stream>>>((float4*)out, n4, counts);
  gate_kernel<<<T_TOK / 8, 256, 0, stream>>>(x, gw, bias, comb, counts, idxList);

  if (preconv) {
    const int wn4 = (int)(wElems / 4);
    convert_bf16_kernel<<<(wn4 + 255) / 256, 256, 0, stream>>>((const float4*)w1, (uint2*)w1b, wn4);
    convert_bf16_kernel<<<(wn4 + 255) / 256, 256, 0, stream>>>((const float4*)w2, (uint2*)w2b, wn4);
    moe_expert_kernel<true><<<dim3(T_TOK / TILE_M, NEXP), 256, 0, stream>>>(
        x, w1b, w2b, w1, w2, comb, counts, idxList, out);
  } else {
    moe_expert_kernel<false><<<dim3(T_TOK / TILE_M, NEXP), 256, 0, stream>>>(
        x, w1b, w2b, w1, w2, comb, counts, idxList, out);
  }
}
